// MaxViTSelfAttention_2259152797914
// MI455X (gfx1250) — matrix operand order repaired and checked by reading, not compiled, not run on hardware
//
#include <hip/hip_runtime.h>

// ---------- types ----------
typedef __attribute__((ext_vector_type(16))) __bf16 v16bf;
typedef __attribute__((ext_vector_type(8)))  float  v8f;
typedef __attribute__((ext_vector_type(4)))  unsigned int u32x4;

struct Frag32 { u32x4 lo, hi; };
static_assert(sizeof(Frag32) == 32, "frag size");

__device__ __forceinline__ v16bf frag_bf16(u32x4 lo, u32x4 hi) {
  Frag32 f; f.lo = lo; f.hi = hi;
  return __builtin_bit_cast(v16bf, f);
}

// A-matrix fragment (16x32 bf16, row-major source).
// Per ISA 7.12.2: lane (M row) holds K = hi*8 + [0,8) and 16 + hi*8 + [0,8).
__device__ __forceinline__ v16bf load_a_frag(const __bf16* rowp, int hi, bool valid) {
  u32x4 z = {0, 0, 0, 0};
  u32x4 lo = z, hq = z;
  if (valid) {
    lo = *(const u32x4*)(rowp + hi * 8);
    hq = *(const u32x4*)(rowp + 16 + hi * 8);
  }
  return frag_bf16(lo, hq);
}

// B-matrix fragment (32x16 bf16) from N-major (transposed) source.
// Lane (N col) holds K = hi*8 + [0,8) and 16 + hi*8 + [0,8).
__device__ __forceinline__ v16bf load_b_frag(const __bf16* colp, int hi, bool valid) {
  u32x4 z = {0, 0, 0, 0};
  u32x4 lo = z, hq = z;
  if (valid) {
    const __bf16* p = colp + hi * 8;
    lo = *(const u32x4*)(p);
    hq = *(const u32x4*)(p + 16);
  }
  return frag_bf16(lo, hq);
}

// ---------- problem constants ----------
#define DIMC   256
#define HEADS  8
#define DHEAD  64
#define INNERC 512
#define NTOK   49
#define NWIN   4096
#define ROWS   (NWIN * NTOK)       // 200704, divisible by 64
#define NQKV   (3 * INNERC)        // 1536
#define KSTEP  64

// ---------- 1. LayerNorm + fp32 -> bf16 ----------
__launch_bounds__(DIMC)
__global__ void ln_to_bf16(const float* __restrict__ x,
                           const float* __restrict__ gamma,
                           const float* __restrict__ beta,
                           __bf16* __restrict__ xn) {
  const int token = blockIdx.x;
  const int t = threadIdx.x;
  __shared__ float red[DIMC];
  float v = x[(size_t)token * DIMC + t];
  red[t] = v;
  __syncthreads();
  for (int s = DIMC / 2; s > 0; s >>= 1) {
    if (t < s) red[t] += red[t + s];
    __syncthreads();
  }
  float mu = red[0] * (1.0f / DIMC);
  __syncthreads();
  float d = v - mu;
  red[t] = d * d;
  __syncthreads();
  for (int s = DIMC / 2; s > 0; s >>= 1) {
    if (t < s) red[t] += red[t + s];
    __syncthreads();
  }
  float rstd = rsqrtf(red[0] * (1.0f / DIMC) + 1e-5f);
  float y = d * rstd * gamma[t] + beta[t];
  xn[(size_t)token * DIMC + t] = (__bf16)y;
}

// ---------- 2. weight transpose + convert: src[K][N] -> dst[N][K] bf16 ----------
__global__ void transpose_to_bf16(const float* __restrict__ src,
                                  __bf16* __restrict__ dst, int K, int N) {
  int idx = blockIdx.x * blockDim.x + threadIdx.x;
  if (idx < K * N) {
    int k = idx / N, n = idx - k * N;
    dst[(size_t)n * K + k] = (__bf16)src[idx];
  }
}

// ---------- shared GEMM core: 64(M) x 128(N) block tile, async-LDS double buffer ----------
// A: [M][KTOT] row-major, rows blockIdx.y*64 .. +64
// BT: [N][KTOT] (i.e. weight transposed), cols blockIdx.x*128 .. +128
// acc[4]: per-wave 16x64 strip of f32 accumulators.
template <int KTOT>
__device__ __forceinline__ void wmma_gemm_64x128(const __bf16* __restrict__ Ag,
                                                 const __bf16* __restrict__ BgT,
                                                 v8f acc[4]) {
  __shared__ __align__(16) __bf16 As[2][64][KSTEP];    // 2 x 8 KB
  __shared__ __align__(16) __bf16 Bs[2][128][KSTEP];   // 2 x 16 KB

  const int tid = threadIdx.x;
  const int rowBlk = blockIdx.y * 64;
  const int colBlk = blockIdx.x * 128;

  // cooperative async stage of one K-slab (CDNA5 ASYNCcnt path)
  auto stage_tiles = [&](int buf, int kk) {
#pragma unroll
    for (int s = 0; s < 2; ++s) {              // A: 64 rows x 128B = 512 x 16B
      int seg = tid + s * 256;
      int r = seg >> 3;
      int c = (seg & 7) * 8;
      unsigned lds = (unsigned)(size_t)&As[buf][r][c];
      unsigned long long ga =
          (unsigned long long)(size_t)(Ag + (size_t)(rowBlk + r) * KTOT + kk + c);
      asm volatile("global_load_async_to_lds_b128 %0, %1, off"
                   :: "v"(lds), "v"(ga) : "memory");
    }
#pragma unroll
    for (int s = 0; s < 4; ++s) {              // B: 128 rows x 128B = 1024 x 16B
      int seg = tid + s * 256;
      int r = seg >> 3;
      int c = (seg & 7) * 8;
      unsigned lds = (unsigned)(size_t)&Bs[buf][r][c];
      unsigned long long ga =
          (unsigned long long)(size_t)(BgT + (size_t)(colBlk + r) * KTOT + kk + c);
      asm volatile("global_load_async_to_lds_b128 %0, %1, off"
                   :: "v"(lds), "v"(ga) : "memory");
    }
  };

  const int wid  = tid >> 5;
  const int lane = tid & 31;
  const int l15  = lane & 15;
  const int hi   = lane >> 4;
  const int wm = (wid & 3) * 16;   // wave row offset inside 64-row tile
  const int wn = (wid >> 2) * 64;  // wave col offset inside 128-col tile

  stage_tiles(0, 0);
  asm volatile("s_wait_asynccnt 0x0" ::: "memory");
  __syncthreads();

  int buf = 0;
  for (int kk = 0; kk < KTOT; kk += KSTEP) {
    if (kk + KSTEP < KTOT) stage_tiles(buf ^ 1, kk + KSTEP);  // overlap next slab
#pragma unroll
    for (int ks = 0; ks < KSTEP; ks += 32) {
      v16bf a = load_a_frag(&As[buf][wm + l15][ks], hi, true);
#pragma unroll
      for (int nt = 0; nt < 4; ++nt) {
        v16bf b = load_b_frag(&Bs[buf][wn + nt * 16 + l15][ks], hi, true);
        acc[nt] = __builtin_amdgcn_wmma_f32_16x16x32_bf16(
            false, a, false, b, (short)0, acc[nt], false, false);
      }
    }
    asm volatile("s_wait_asynccnt 0x0" ::: "memory");
    __syncthreads();
    buf ^= 1;
  }
}

// ---------- 3. QKV GEMM: xn(ROWS x 256) @ Wqkv(256 x 1536) ----------
__launch_bounds__(256)
__global__ void qkv_gemm(const __bf16* __restrict__ xn,
                         const __bf16* __restrict__ wT,   // [1536][256]
                         __bf16* __restrict__ qb,
                         __bf16* __restrict__ kb,
                         __bf16* __restrict__ vb) {
  const int wid  = threadIdx.x >> 5;
  const int lane = threadIdx.x & 31;
  const int l15  = lane & 15;
  const int hi   = lane >> 4;
  const int rowBase = blockIdx.y * 64 + (wid & 3) * 16;
  const int colBase = blockIdx.x * 128 + (wid >> 2) * 64;

  v8f z = {};
  v8f acc[4] = {z, z, z, z};
  wmma_gemm_64x128<DIMC>(xn, wT, acc);

#pragma unroll
  for (int nt = 0; nt < 4; ++nt) {
#pragma unroll
    for (int r = 0; r < 8; ++r) {
      int i = rowBase + r + hi * 8;           // flat token row
      int c = colBase + nt * 16 + l15;        // column in [0,1536)
      float val = acc[nt][r];
      int which = c >> 9;                      // 0=q 1=k 2=v
      int inner = c & 511;
      int h = inner >> 6;
      int d = inner & 63;
      int bw = i / NTOK;
      int n  = i - bw * NTOK;
      size_t off = ((size_t)(bw * HEADS + h) * NTOK + n) * DHEAD + d;
      if (which == 0)      qb[off] = (__bf16)(val * 0.125f);  // pre-scale q
      else if (which == 1) kb[off] = (__bf16)val;
      else                 vb[off] = (__bf16)val;
    }
  }
}

// ---------- 4. attention per (b,h): sim -> +bias -> softmax -> P@V ----------
__launch_bounds__(128)
__global__ void attn_kernel(const __bf16* __restrict__ qb,
                            const __bf16* __restrict__ kb,
                            const __bf16* __restrict__ vb,
                            const float* __restrict__ bias_table, // [169][8]
                            const int* __restrict__ rel_idx,      // [49][49]
                            __bf16* __restrict__ aout) {          // [ROWS][512]
  const int bh = blockIdx.x;
  const int h  = bh & 7;
  const int bw = bh >> 3;
  const __bf16* q = qb + (size_t)bh * NTOK * DHEAD;
  const __bf16* k = kb + (size_t)bh * NTOK * DHEAD;
  const __bf16* v = vb + (size_t)bh * NTOK * DHEAD;

  __shared__ __align__(16) float  sim[64 * 64];
  __shared__ __align__(16) __bf16 probs[64 * 64];
  __shared__ __align__(16) __bf16 vT[64 * 64];

  const int tid  = threadIdx.x;
  const int wid  = tid >> 5;
  const int lane = tid & 31;
  const int l15  = lane & 15;
  const int hi   = lane >> 4;

  // stage v^T into LDS (zero-pad tokens >= 49)
  for (int idx = tid; idx < 64 * 64; idx += 128) {
    int j = idx >> 6, d = idx & 63;
    vT[d * 64 + j] = (j < NTOK) ? v[j * DHEAD + d] : (__bf16)0.0f;
  }

  // ---- sim = q @ k^T (64x64 padded), wave w owns rows [w*16, w*16+16) ----
  const int rowBase = wid * 16;
  const int mrow = rowBase + l15;
  const bool mvalid = mrow < NTOK;
  v8f z = {};
  v8f acc[4] = {z, z, z, z};
  for (int kk = 0; kk < DHEAD; kk += 32) {
    v16bf a = load_a_frag(q + (size_t)mrow * DHEAD + kk, hi, mvalid);
#pragma unroll
    for (int nt = 0; nt < 4; ++nt) {
      int j = nt * 16 + l15;
      v16bf b = load_b_frag(k + (size_t)j * DHEAD + kk, hi, j < NTOK);
      acc[nt] = __builtin_amdgcn_wmma_f32_16x16x32_bf16(
          false, a, false, b, (short)0, acc[nt], false, false);
    }
  }
  // add relative-position bias, mask, spill to LDS
#pragma unroll
  for (int nt = 0; nt < 4; ++nt) {
#pragma unroll
    for (int r = 0; r < 8; ++r) {
      int i = rowBase + r + hi * 8;
      int j = nt * 16 + l15;
      float val = acc[nt][r];
      if (j < NTOK) {
        if (i < NTOK) val += bias_table[rel_idx[i * NTOK + j] * HEADS + h];
      } else {
        val = -3.0e38f;
      }
      sim[i * 64 + j] = val;
    }
  }
  __syncthreads();

  // ---- softmax: one thread per row ----
  if (tid < 64) {
    int i = tid;
    float m = -3.0e38f;
    for (int j = 0; j < NTOK; ++j) m = fmaxf(m, sim[i * 64 + j]);
    float s = 0.0f;
    for (int j = 0; j < NTOK; ++j) s += __expf(sim[i * 64 + j] - m);
    float inv = 1.0f / s;
    for (int j = 0; j < 64; ++j) {
      float p = (j < NTOK) ? __expf(sim[i * 64 + j] - m) * inv : 0.0f;
      probs[i * 64 + j] = (__bf16)p;
    }
  }
  __syncthreads();

  // ---- out = P @ V : fragments from LDS ----
  v8f oacc[4] = {z, z, z, z};
  for (int kk = 0; kk < 64; kk += 32) {
    v16bf a = load_a_frag(&probs[(size_t)mrow * 64 + kk], hi, true);
#pragma unroll
    for (int nt = 0; nt < 4; ++nt) {
      int d = nt * 16 + l15;
      v16bf b = load_b_frag(&vT[(size_t)d * 64 + kk], hi, true);
      oacc[nt] = __builtin_amdgcn_wmma_f32_16x16x32_bf16(
          false, a, false, b, (short)0, oacc[nt], false, false);
    }
  }
#pragma unroll
  for (int nt = 0; nt < 4; ++nt) {
#pragma unroll
    for (int r = 0; r < 8; ++r) {
      int i = rowBase + r + hi * 8;
      if (i < NTOK) {
        int d = nt * 16 + l15;
        aout[((size_t)bw * NTOK + i) * INNERC + h * DHEAD + d] = (__bf16)oacc[nt][r];
      }
    }
  }
}

// ---------- 5. output GEMM: aout(ROWS x 512) @ Wout(512 x 256) + b_out ----------
__launch_bounds__(256)
__global__ void out_gemm(const __bf16* __restrict__ ain,
                         const __bf16* __restrict__ wT,   // [256][512]
                         const float* __restrict__ bout,
                         float* __restrict__ out) {
  const int wid  = threadIdx.x >> 5;
  const int lane = threadIdx.x & 31;
  const int l15  = lane & 15;
  const int hi   = lane >> 4;
  const int rowBase = blockIdx.y * 64 + (wid & 3) * 16;
  const int colBase = blockIdx.x * 128 + (wid >> 2) * 64;

  v8f z = {};
  v8f acc[4] = {z, z, z, z};
  wmma_gemm_64x128<INNERC>(ain, wT, acc);

#pragma unroll
  for (int nt = 0; nt < 4; ++nt) {
#pragma unroll
    for (int r = 0; r < 8; ++r) {
      int i = rowBase + r + hi * 8;
      int c = colBase + nt * 16 + l15;
      out[(size_t)i * DIMC + c] = acc[nt][r] + bout[c];
    }
  }
}

// ---------- launcher ----------
extern "C" void kernel_launch(void* const* d_in, const int* in_sizes, int n_in,
                              void* d_out, int out_size, void* d_ws, size_t ws_size,
                              hipStream_t stream) {
  const float* x          = (const float*)d_in[0];
  const float* gamma      = (const float*)d_in[1];
  const float* beta       = (const float*)d_in[2];
  const float* W_qkv      = (const float*)d_in[3];
  const float* W_out      = (const float*)d_in[4];
  const float* b_out      = (const float*)d_in[5];
  const float* bias_table = (const float*)d_in[6];
  const int*   rel_idx    = (const int*)d_in[7];
  float* out = (float*)d_out;

  char* ws = (char*)d_ws;
  size_t off = 0;
  auto alloc = [&](size_t bytes) -> void* {
    void* p = ws + off;
    off += (bytes + 255) & ~(size_t)255;
    return p;
  };
  __bf16* xn     = (__bf16*)alloc((size_t)ROWS * DIMC * 2);      // 103 MB
  __bf16* wqkvT  = (__bf16*)alloc((size_t)NQKV * DIMC * 2);      // 0.8 MB
  __bf16* woutT  = (__bf16*)alloc((size_t)DIMC * INNERC * 2);    // 0.25 MB
  __bf16* qbuf   = (__bf16*)alloc((size_t)NWIN * HEADS * NTOK * DHEAD * 2); // 205 MB
  __bf16* kbuf   = (__bf16*)alloc((size_t)NWIN * HEADS * NTOK * DHEAD * 2);
  __bf16* vbuf   = (__bf16*)alloc((size_t)NWIN * HEADS * NTOK * DHEAD * 2);
  __bf16* aout   = (__bf16*)alloc((size_t)ROWS * INNERC * 2);    // 205 MB
  (void)ws_size; (void)in_sizes; (void)n_in; (void)out_size;

  ln_to_bf16<<<ROWS, DIMC, 0, stream>>>(x, gamma, beta, xn);
  transpose_to_bf16<<<(DIMC * NQKV + 255) / 256, 256, 0, stream>>>(W_qkv, wqkvT, DIMC, NQKV);
  transpose_to_bf16<<<(INNERC * DIMC + 255) / 256, 256, 0, stream>>>(W_out, woutT, INNERC, DIMC);

  dim3 g1(NQKV / 128, ROWS / 64);   // (12, 3136)
  qkv_gemm<<<g1, 256, 0, stream>>>(xn, wqkvT, qbuf, kbuf, vbuf);

  attn_kernel<<<NWIN * HEADS, 128, 0, stream>>>(qbuf, kbuf, vbuf, bias_table, rel_idx, aout);

  dim3 g2(DIMC / 128, ROWS / 64);   // (2, 3136)
  out_gemm<<<g2, 256, 0, stream>>>(aout, woutT, b_out, out);
}
